// VAE_Hist_GLM2_18940805775498
// MI455X (gfx1250) — compile-verified, matrix operand order checked by reading, not run on hardware
//
#include <hip/hip_runtime.h>
#include <hip/hip_bf16.h>
#include <math.h>

// ---------------------------------------------------------------------------
// Problem constants (match reference)
// ---------------------------------------------------------------------------
#define SUB_NO 20
#define T_NO   401
#define N_E    2000
#define N_I    200
#define T_DATA 50000
#define EXP_NO 81
#define HID_NO 6
#define CT     256            // time-tile for LDS conv kernels

#define NCH_E  ((N_E + 31) / 32)   // 63 K-chunks for e-projection
#define NCH_I  ((N_I + 31) / 32)   // 7 K-chunks for i-projection

// Workspace layout (float offsets), every offset a multiple of 16 floats (64B)
// so float4 / v16h accesses stay naturally aligned. Total ~18.5 MB.
#define OFF_EKER 0                                   // 20*401 -> 8032
#define OFF_IKER 8032
#define OFF_SPK  16064                               // 401 -> 416
#define OFF_RK1  16480                               // 2406 -> 2416
#define OFF_RK2  18896                               // 2406 -> 2416
#define OFF_BPE  21312                               // 63*2*32*16 halves = 32256 floats
#define OFF_BPI  53568                               // 7*2*32*16 halves  = 3584 floats
#define OFF_SYNE 57152                               // 32*T  (padded SoA)
#define OFF_SYNI (OFF_SYNE + 32*T_DATA)
#define OFF_SYNF (OFF_SYNI + 32*T_DATA)              // 20*T
#define OFF_H0   (OFF_SYNF + SUB_NO*T_DATA)          // T -> 50016
#define OFF_H1   (OFF_H0 + 50016)                    // 6*T

typedef __attribute__((ext_vector_type(16))) _Float16 v16h;
typedef __attribute__((ext_vector_type(8)))  float    v8f;

// 16-bit WMMA A/B layout: element e of the per-lane v16h holds
//   K = e + (e>=8 ? 8 : 0) + (lane>=16 ? 8 : 0)   within the 32-wide K chunk.
__device__ __forceinline__ int wmma_klocal(int e, int half) {
  return e + ((e >= 8) ? 8 : 0) + half * 8;
}

__device__ __forceinline__ v16h cvt_a16(float4 a0, float4 a1, float4 a2, float4 a3) {
  v16h a;
  a[0]  = (_Float16)a0.x; a[1]  = (_Float16)a0.y; a[2]  = (_Float16)a0.z; a[3]  = (_Float16)a0.w;
  a[4]  = (_Float16)a1.x; a[5]  = (_Float16)a1.y; a[6]  = (_Float16)a1.z; a[7]  = (_Float16)a1.w;
  a[8]  = (_Float16)a2.x; a[9]  = (_Float16)a2.y; a[10] = (_Float16)a2.z; a[11] = (_Float16)a2.w;
  a[12] = (_Float16)a3.x; a[13] = (_Float16)a3.y; a[14] = (_Float16)a3.z; a[15] = (_Float16)a3.w;
  return a;
}

// ---------------------------------------------------------------------------
// Kernel 1: build temporal kernels + out_filters
// ---------------------------------------------------------------------------
__global__ void build_kernels_k(const float* __restrict__ W_syn,
                                const float* __restrict__ Tau_syn,
                                const float* __restrict__ Delta_syn,
                                const float* __restrict__ W_spk,
                                const float* __restrict__ Tau_spk,
                                const float* __restrict__ W_root_1,
                                const float* __restrict__ W_root_2,
                                const float* __restrict__ exp_basis,
                                float* __restrict__ eker,
                                float* __restrict__ iker,
                                float* __restrict__ spk,
                                float* __restrict__ rk1,
                                float* __restrict__ rk2,
                                float* __restrict__ out_filters) {
  const int NEI = SUB_NO * T_NO;
  int id = blockIdx.x * blockDim.x + threadIdx.x;
  if (id < NEI) {
    int s = id / T_NO, j = id % T_NO;
    float de = Delta_syn[s*2+0], di = Delta_syn[s*2+1];
    float taue = Tau_syn[s*2+0], taui = Tau_syn[s*2+1];
    float we = W_syn[s*2+0],     wi = W_syn[s*2+1];
    float te = fmaxf((float)j - de, 0.f);
    float ti = fmaxf((float)j - di, 0.f);
    float tte = te / (taue*taue);
    float tti = ti / (taui*taui);
    float ev =  tte * expf(-tte) * we*we;
    float iv = -(tti * expf(-tti) * wi*wi);
    eker[id] = ev;
    iker[id] = iv;
    out_filters[id]       = ev;   // full_e_kern rows 0..19
    out_filters[NEI + id] = iv;   // full_i_kern rows 20..39
  } else if (id < NEI + T_NO) {
    int j = id - NEI;
    float tau = Tau_spk[0];
    float tt = (float)j / (tau*tau);
    spk[j] = tt * expf(-tt) * W_spk[0]*W_spk[0];
  } else if (id < NEI + T_NO + HID_NO*T_NO) {
    int r = id - NEI - T_NO;
    int c = r / T_NO, j = r % T_NO;
    float acc = 0.f;
    #pragma unroll 3
    for (int e = 0; e < EXP_NO; ++e)
      acc = fmaf(W_root_1[c*EXP_NO + e], exp_basis[e*T_NO + j], acc);
    rk1[r] = acc;
  } else if (id < NEI + T_NO + 2*HID_NO*T_NO) {
    int r = id - NEI - T_NO - HID_NO*T_NO;
    int c = r / T_NO, j = r % T_NO;
    float acc = 0.f;
    #pragma unroll 3
    for (int e = 0; e < EXP_NO; ++e)
      acc = fmaf(W_root_2[c*EXP_NO + e], exp_basis[e*T_NO + j], acc);
    rk2[r] = acc;
  }
}

// ---------------------------------------------------------------------------
// Kernel 2: pack B = Csyn^T into per-lane v16h WMMA layout, f16, zero-padded.
// Layout: [chunk][ntile(2)][lane(32)] x 16 halves.
// ---------------------------------------------------------------------------
__global__ void pack_b_k(const float* __restrict__ Csyn,  // [N x K] row-major
                         _Float16* __restrict__ Bpack,
                         int K, int N, int nChunks) {
  int id = blockIdx.x * blockDim.x + threadIdx.x;
  if (id >= nChunks * 2 * 32) return;
  int lane  = id & 31;
  int nt    = (id >> 5) & 1;
  int chunk = id >> 6;
  int half  = lane >> 4;
  int n     = (lane & 15) + nt * 16;
  _Float16* dst = Bpack + (size_t)id * 16;
  #pragma unroll
  for (int e = 0; e < 16; ++e) {
    int k = chunk * 32 + wmma_klocal(e, half);
    float v = (k < K && n < N) ? Csyn[(size_t)n * K + k] : 0.f;
    dst[e] = (_Float16)v;
  }
}

// ---------------------------------------------------------------------------
// Kernel 3: WMMA GEMM   out[n][t] = sum_k A[t,k] * Csyn[n,k]
// One wave32 per block; 16(t) x 32(n) tile; f32->f16 convert in registers,
// f32 accumulate. Grid must be exactly T/16 blocks (T % 16 == 0).
// Main K loop: unconditional b128 loads. K tail: clamped address + cndmask.
// ---------------------------------------------------------------------------
__global__ void syn_gemm_wmma_k(const float* __restrict__ A,        // [T x K]
                                const _Float16* __restrict__ Bpack, // packed
                                float* __restrict__ outSoA,         // [32][T]
                                int T, int K) {
  const int t0   = blockIdx.x * 16;
  const int lane = threadIdx.x;   // 0..31
  const int half = lane >> 4;
  const int m    = lane & 15;
  const float* arow = A + (size_t)(t0 + m) * K;
  const v16h*  bp   = (const v16h*)Bpack;

  v8f acc0 = {};
  v8f acc1 = {};

  const int KMain = K & ~31;
  int kb = 0;
  for (; kb < KMain; kb += 32) {
    const float4* p0 = (const float4*)(arow + kb + half * 8);
    const float4* p1 = (const float4*)(arow + kb + 16 + half * 8);
    float4 a0 = p0[0], a1 = p0[1];
    float4 a2 = p1[0], a3 = p1[1];
    v16h a = cvt_a16(a0, a1, a2, a3);
    size_t bbase = (size_t)(kb >> 5) * 64;           // chunk*2*32 v16h entries
    v16h b0 = bp[bbase + lane];
    v16h b1 = bp[bbase + 32 + lane];
    acc0 = __builtin_amdgcn_wmma_f32_16x16x32_f16(false, a, false, b0,
                                                  (short)0, acc0, false, false);
    acc1 = __builtin_amdgcn_wmma_f32_16x16x32_f16(false, a, false, b1,
                                                  (short)0, acc1, false, false);
    __builtin_prefetch(arow + kb + 64, 0, 1);        // speculative; OOB dropped
  }
  if (kb < K) {                                      // uniform tail branch
    v16h a;
    #pragma unroll
    for (int e = 0; e < 16; ++e) {
      int k  = kb + wmma_klocal(e, half);
      int kc = (k < K) ? k : (K - 1);                // clamped address: no branch
      float v = arow[kc];
      a[e] = (_Float16)((k < K) ? v : 0.f);          // v_cndmask mask
    }
    size_t bbase = (size_t)(kb >> 5) * 64;
    v16h b0 = bp[bbase + lane];
    v16h b1 = bp[bbase + 32 + lane];
    acc0 = __builtin_amdgcn_wmma_f32_16x16x32_f16(false, a, false, b0,
                                                  (short)0, acc0, false, false);
    acc1 = __builtin_amdgcn_wmma_f32_16x16x32_f16(false, a, false, b1,
                                                  (short)0, acc1, false, false);
  }

  // D layout: lane l owns column n = (l&15)(+16), rows t0+8*half .. t0+8*half+7
  // -> two contiguous float4 stores per accumulator, unconditional (SoA is
  //    padded to 32 rows; downstream only reads rows 0..19).
  float* c0 = outSoA + (size_t)m * T + t0 + 8 * half;
  float4 s00 = make_float4(acc0[0], acc0[1], acc0[2], acc0[3]);
  float4 s01 = make_float4(acc0[4], acc0[5], acc0[6], acc0[7]);
  *(float4*)(c0)     = s00;
  *(float4*)(c0 + 4) = s01;
  float* c1 = outSoA + (size_t)(m + 16) * T + t0 + 8 * half;
  float4 s10 = make_float4(acc1[0], acc1[1], acc1[2], acc1[3]);
  float4 s11 = make_float4(acc1[4], acc1[5], acc1[6], acc1[7]);
  *(float4*)(c1)     = s10;
  *(float4*)(c1 + 4) = s11;
}

// ---------------------------------------------------------------------------
// Kernel 4: depthwise causal 401-tap conv, e and i fused:
//   syn[s][t] = sum_j eker[s,j]*syn_e[s][t-j] + iker[s,j]*syn_i[s][t-j]
// ---------------------------------------------------------------------------
__global__ void dw_conv_k(const float* __restrict__ se,   // [32][T] padded SoA
                          const float* __restrict__ si,
                          const float* __restrict__ eker, // [SUB][401]
                          const float* __restrict__ iker,
                          float* __restrict__ syn,        // [SUB][T]
                          int T) {
  __shared__ float ebuf[CT + T_NO - 1];
  __shared__ float ibuf[CT + T_NO - 1];
  __shared__ float ek[T_NO];
  __shared__ float ik[T_NO];
  const int s   = blockIdx.y;
  const int t0  = blockIdx.x * CT;
  const int tid = threadIdx.x;
  const float* ecol = se + (size_t)s * T;
  const float* icol = si + (size_t)s * T;
  for (int x = tid; x < CT + T_NO - 1; x += CT) {
    int t = t0 - (T_NO - 1) + x;
    bool ok = (t >= 0 && t < T);
    ebuf[x] = ok ? ecol[t] : 0.f;
    ibuf[x] = ok ? icol[t] : 0.f;
  }
  for (int j = tid; j < T_NO; j += CT) {
    ek[j] = eker[s*T_NO + j];
    ik[j] = iker[s*T_NO + j];
  }
  __syncthreads();
  int t = t0 + tid;
  if (t < T) {
    float acc = 0.f;
    #pragma unroll 4
    for (int j = 0; j < T_NO; ++j)
      acc = fmaf(ek[j], ebuf[tid + 400 - j], fmaf(ik[j], ibuf[tid + 400 - j], acc));
    syn[(size_t)s * T + t] = acc;
  }
}

// ---------------------------------------------------------------------------
// Kernel 5: subunit tree recursion -> h0 = tanh(root input)
// ---------------------------------------------------------------------------
__global__ void tree_k(const float* __restrict__ syn,   // [SUB][T]
                       const float* __restrict__ C_den, // [SUB][SUB]
                       const float* __restrict__ W_sub, // [SUB]
                       const float* __restrict__ Theta, // [SUB]
                       float* __restrict__ h0, int T) {
  int t = blockIdx.x * blockDim.x + threadIdx.x;
  if (t >= T) return;
  float w2[SUB_NO];
  #pragma unroll
  for (int j = 0; j < SUB_NO; ++j) w2[j] = W_sub[j] * W_sub[j];
  float ns[SUB_NO];
  #pragma unroll
  for (int j = 0; j < SUB_NO; ++j) ns[j] = 0.f;
  #pragma unroll
  for (int i = SUB_NO - 1; i >= 1; --i) {
    float child = 0.f;
    #pragma unroll
    for (int j = 0; j < SUB_NO; ++j)
      child = fmaf(C_den[i*SUB_NO + j] * w2[j], ns[j], child);
    ns[i] = tanhf(syn[(size_t)i * T + t] + Theta[i] + child);
  }
  float child = 0.f;
  #pragma unroll
  for (int j = 0; j < SUB_NO; ++j)
    child = fmaf(C_den[j] * w2[j], ns[j], child);
  h0[t] = tanhf(syn[t] + Theta[0] + child);
}

// ---------------------------------------------------------------------------
// Kernel 6: root conv layer 1 ("same", pad 200):
//   h1[c][t] = tanh( sum_j rk1[c,j] * h0[t + 200 - j] )
// ---------------------------------------------------------------------------
__global__ void root1_k(const float* __restrict__ h0,
                        const float* __restrict__ rk1,
                        float* __restrict__ h1, int T) {
  __shared__ float hbuf[CT + T_NO - 1];
  __shared__ float kb[T_NO];
  const int c = blockIdx.y;
  const int t0 = blockIdx.x * CT;
  const int tid = threadIdx.x;
  for (int x = tid; x < CT + T_NO - 1; x += CT) {
    int t = t0 - 200 + x;
    hbuf[x] = (t >= 0 && t < T) ? h0[t] : 0.f;
  }
  for (int j = tid; j < T_NO; j += CT) kb[j] = rk1[c*T_NO + j];
  __syncthreads();
  int t = t0 + tid;
  if (t < T) {
    float acc = 0.f;
    #pragma unroll 4
    for (int j = 0; j < T_NO; ++j)
      acc = fmaf(kb[j], hbuf[tid + 400 - j], acc);
    h1[(size_t)c * T + t] = tanhf(acc);
  }
}

// ---------------------------------------------------------------------------
// Kernel 7: root conv layer 2 + sigmoid -> final_Z
// ---------------------------------------------------------------------------
__global__ void root2_k(const float* __restrict__ h1,
                        const float* __restrict__ rk2,
                        float* __restrict__ Zout, int T) {
  __shared__ float hbuf[HID_NO][CT + T_NO - 1];
  __shared__ float kb[HID_NO][T_NO];
  const int t0 = blockIdx.x * CT;
  const int tid = threadIdx.x;
  for (int c = 0; c < HID_NO; ++c) {
    for (int x = tid; x < CT + T_NO - 1; x += CT) {
      int t = t0 - 200 + x;
      hbuf[c][x] = (t >= 0 && t < T) ? h1[(size_t)c * T + t] : 0.f;
    }
    for (int j = tid; j < T_NO; j += CT) kb[c][j] = rk2[c*T_NO + j];
  }
  __syncthreads();
  int t = t0 + tid;
  if (t < T) {
    float acc = 0.f;
    for (int c = 0; c < HID_NO; ++c) {
      #pragma unroll 4
      for (int j = 0; j < T_NO; ++j)
        acc = fmaf(kb[c][j], hbuf[c][tid + 400 - j], acc);
    }
    Zout[t] = 1.f / (1.f + expf(-acc));
  }
}

// ---------------------------------------------------------------------------
// Kernel 8: spike-history filter:  V[t] = sum_j spk[j] * Z[t - 1 - j]
// ---------------------------------------------------------------------------
__global__ void spike_hist_k(const float* __restrict__ Z,
                             const float* __restrict__ spk,
                             float* __restrict__ V, int T) {
  __shared__ float zbuf[CT + T_NO];
  __shared__ float kb[T_NO];
  const int t0 = blockIdx.x * CT;
  const int tid = threadIdx.x;
  for (int x = tid; x < CT + T_NO; x += CT) {
    int t = t0 - T_NO + x;
    zbuf[x] = (t >= 0 && t < T) ? Z[t] : 0.f;
  }
  for (int j = tid; j < T_NO; j += CT) kb[j] = spk[j];
  __syncthreads();
  int t = t0 + tid;
  if (t < T) {
    float acc = 0.f;
    #pragma unroll 4
    for (int j = 0; j < T_NO; ++j)
      acc = fmaf(kb[j], zbuf[tid + 400 - j], acc);   // zbuf[tid+400-j] = Z[t-1-j]
    V[t] = acc;
  }
}

// ---------------------------------------------------------------------------
// Host launcher
// ---------------------------------------------------------------------------
extern "C" void kernel_launch(void* const* d_in, const int* in_sizes, int n_in,
                              void* d_out, int out_size, void* d_ws, size_t ws_size,
                              hipStream_t stream) {
  const float* S_e     = (const float*)d_in[0];
  const float* S_i     = (const float*)d_in[1];
  const float* Z       = (const float*)d_in[2];
  const float* W_syn   = (const float*)d_in[3];
  const float* Tau_syn = (const float*)d_in[4];
  const float* Delta   = (const float*)d_in[5];
  const float* W_sub   = (const float*)d_in[6];
  const float* W_spk   = (const float*)d_in[7];
  const float* Tau_spk = (const float*)d_in[8];
  const float* Theta   = (const float*)d_in[9];
  const float* W_root1 = (const float*)d_in[10];
  const float* W_root2 = (const float*)d_in[11];
  const float* C_den   = (const float*)d_in[12];
  const float* C_syn_e = (const float*)d_in[13];
  const float* C_syn_i = (const float*)d_in[14];
  const float* basis   = (const float*)d_in[15];

  float* out = (float*)d_out;   // [0,T): final_V, [T,2T): final_Z, [2T,..): filters
  float* w   = (float*)d_ws;    // ~18.5 MB scratch

  float*     eker  = w + OFF_EKER;
  float*     iker  = w + OFF_IKER;
  float*     spk   = w + OFF_SPK;
  float*     rk1   = w + OFF_RK1;
  float*     rk2   = w + OFF_RK2;
  _Float16*  bpe   = (_Float16*)(w + OFF_BPE);
  _Float16*  bpi   = (_Float16*)(w + OFF_BPI);
  float*     syn_e = w + OFF_SYNE;
  float*     syn_i = w + OFF_SYNI;
  float*     synf  = w + OFF_SYNF;
  float*     h0    = w + OFF_H0;
  float*     h1    = w + OFF_H1;

  const int T = T_DATA;
  const int nTT = (T + CT - 1) / CT;               // 196 time tiles

  // 1) temporal kernels + out_filters, and B-matrix packing
  {
    int total = SUB_NO*T_NO + T_NO + 2*HID_NO*T_NO;
    build_kernels_k<<<(total + 255)/256, 256, 0, stream>>>(
        W_syn, Tau_syn, Delta, W_spk, Tau_spk, W_root1, W_root2, basis,
        eker, iker, spk, rk1, rk2, out + 2*T);
  }
  pack_b_k<<<(NCH_E*64 + 255)/256, 256, 0, stream>>>(C_syn_e, bpe, N_E, SUB_NO, NCH_E);
  pack_b_k<<<(NCH_I*64 + 255)/256, 256, 0, stream>>>(C_syn_i, bpi, N_I, SUB_NO, NCH_I);

  // 2) WMMA synapse projections (memory-bound: ~440 MB @ 23.3 TB/s floor)
  //    T % 16 == 0, so exactly T/16 wave-tiles, no row guards.
  syn_gemm_wmma_k<<<T/16, 32, 0, stream>>>(S_e, bpe, syn_e, T, N_E);
  syn_gemm_wmma_k<<<T/16, 32, 0, stream>>>(S_i, bpi, syn_i, T, N_I);

  // 3) depthwise causal filtering (e + i fused)
  dw_conv_k<<<dim3(nTT, SUB_NO), CT, 0, stream>>>(syn_e, syn_i, eker, iker, synf, T);

  // 4) subunit tree recursion -> h0
  tree_k<<<nTT, CT, 0, stream>>>(synf, C_den, W_sub, Theta, h0, T);

  // 5) root conv head
  root1_k<<<dim3(nTT, HID_NO), CT, 0, stream>>>(h0, rk1, h1, T);
  root2_k<<<nTT, CT, 0, stream>>>(h1, rk2, out + T, T);

  // 6) spike-history filter -> final_V
  spike_hist_k<<<nTT, CT, 0, stream>>>(Z, spk, out, T);
}
